// GAT_2362232013429
// MI455X (gfx1250) — matrix operand order repaired and checked by reading, not compiled, not run on hardware
//
#include <hip/hip_runtime.h>

typedef __attribute__((ext_vector_type(16))) _Float16 v16h;
typedef __attribute__((ext_vector_type(8)))  _Float16 v8h;
typedef __attribute__((ext_vector_type(8)))  float    v8f;

#define NN   100000      // nodes
#define NE   1600000     // edges
#define KDIM 128         // input dim of both GEMMs
#define HID  32
#define HEADS 4
#define NCLS 40
#define NCLS_PAD 48

// ---------- helpers ----------
__device__ __forceinline__ unsigned fenc(float f) {
    unsigned u = __float_as_uint(f);
    return (u & 0x80000000u) ? ~u : (u | 0x80000000u);   // order-preserving map
}
__device__ __forceinline__ float fdec(unsigned k) {
    unsigned u = (k & 0x80000000u) ? (k & 0x7FFFFFFFu) : ~k;
    return __uint_as_float(u);
}
__device__ __forceinline__ float lrelu(float v) { return v > 0.f ? v : 0.2f * v; }

__global__ void zero_u32(unsigned* __restrict__ p, unsigned n) {
    unsigned i = blockIdx.x * blockDim.x + threadIdx.x;
    if (i < n) p[i] = 0u;
}

__global__ void cvt_f32_to_f16(const float* __restrict__ x, _Float16* __restrict__ y, unsigned n) {
    unsigned i = blockIdx.x * blockDim.x + threadIdx.x;
    if (i < n) y[i] = (_Float16)x[i];
}

// relu(out1 + b1) -> fp16 input of layer 2
__global__ void relu_bias_cvt(const float* __restrict__ h, const float* __restrict__ b,
                              _Float16* __restrict__ y, unsigned n) {
    unsigned i = blockIdx.x * blockDim.x + threadIdx.x;
    if (i >= n) return;
    float v = h[i] + b[i % KDIM];
    y[i] = (_Float16)(v > 0.0f ? v : 0.0f);
}

__global__ void init_out_bias(float* __restrict__ out, const float* __restrict__ b, unsigned n) {
    unsigned i = blockIdx.x * blockDim.x + threadIdx.x;
    if (i < n) out[i] = b[i % NCLS];
}

// Pack W[K x ncols] (f32, row-major) into per-lane WMMA B fragments (f16).
// frag[((nt*4+kb)*32+lane)*16 + e] = W[kb*32 + (lane>=16?8:0) + e + (e>=8?8:0)][nt*16 + lane%16]
__global__ void pack_w(const float* __restrict__ W, _Float16* __restrict__ Frag,
                       int ncols, int n_tiles) {
    int idx = blockIdx.x * blockDim.x + threadIdx.x;
    int total = n_tiles * 4 * 32 * 16;
    if (idx >= total) return;
    int e    = idx & 15;
    int lane = (idx >> 4) & 31;
    int kb   = (idx >> 9) & 3;
    int nt   = idx >> 11;
    int K    = kb * 32 + ((lane & 16) ? 8 : 0) + e + ((e & 8) ? 8 : 0);
    int col  = nt * 16 + (lane & 15);
    float v  = (col < ncols) ? W[K * ncols + col] : 0.0f;
    Frag[idx] = (_Float16)v;
}

// C[M x out_ld(cols n_tiles*16)] = X[M x 128](f16) * Bfrag ; one 16x16 tile per wave
__global__ void gemm_wmma(const _Float16* __restrict__ X, const _Float16* __restrict__ Bfrag,
                          float* __restrict__ Out, int m_tiles, int n_tiles, int out_ld) {
    int wave_id = blockIdx.x * (blockDim.x >> 5) + (threadIdx.x >> 5);
    int lane    = threadIdx.x & 31;
    if (wave_id >= m_tiles * n_tiles) return;       // uniform per wave
    int mt = wave_id / n_tiles, nt = wave_id % n_tiles;

    const _Float16* xrow = X + (size_t)(mt * 16 + (lane & 15)) * KDIM;
    int khalf = (lane >> 4) << 3;                   // 0 or 8

    v8f acc = {};
#pragma unroll
    for (int kb = 0; kb < 4; ++kb) {
        int k0 = kb * 32 + khalf;
        union { v16h v; struct { v8h lo, hi; } p; } a;
        a.p.lo = *(const v8h*)(xrow + k0);          // K = k0 .. k0+7
        a.p.hi = *(const v8h*)(xrow + k0 + 16);     // K = k0+16 .. k0+23
        v16h b = *(const v16h*)(Bfrag + ((size_t)(nt * 4 + kb) * 32 + lane) * 16);
        acc = __builtin_amdgcn_wmma_f32_16x16x32_f16(false, a.v, false, b,
                                                     (short)0, acc, false, false);
    }
    int col   = nt * 16 + (lane & 15);
    int rbase = mt * 16 + ((lane >> 4) << 3);
#pragma unroll
    for (int v = 0; v < 8; ++v)
        Out[(size_t)(rbase + v) * out_ld + col] = acc[v];
}

// alpha_src/alpha_dst, layer 1: one thread per (node, head); float4 inner loads
__global__ void alpha1_kernel(const float* __restrict__ h1, const float* __restrict__ asrc,
                              const float* __restrict__ adst,
                              float* __restrict__ as_out, float* __restrict__ ad_out) {
    int tid = blockIdx.x * blockDim.x + threadIdx.x;
    if (tid >= NN * HEADS) return;
    int n = tid / HEADS, hd = tid % HEADS;
    const float4* hp = (const float4*)(h1 + (size_t)n * KDIM + hd * HID);
    const float4* ap = (const float4*)(asrc + hd * HID);
    const float4* bp = (const float4*)(adst + hd * HID);
    float s = 0.f, t = 0.f;
#pragma unroll
    for (int f = 0; f < HID / 4; ++f) {
        float4 v = hp[f], a = ap[f], b = bp[f];
        s += v.x * a.x + v.y * a.y + v.z * a.z + v.w * a.w;
        t += v.x * b.x + v.y * b.y + v.z * b.z + v.w * b.w;
    }
    as_out[tid] = s; ad_out[tid] = t;
}

// alpha, layer 2: one thread per node
__global__ void alpha2_kernel(const float* __restrict__ h2, const float* __restrict__ asrc,
                              const float* __restrict__ adst,
                              float* __restrict__ as_out, float* __restrict__ ad_out) {
    int n = blockIdx.x * blockDim.x + threadIdx.x;
    if (n >= NN) return;
    const float4* hp = (const float4*)(h2 + (size_t)n * NCLS_PAD);
    float s = 0.f, t = 0.f;
#pragma unroll
    for (int f4 = 0; f4 < NCLS / 4; ++f4) {
        float4 v = hp[f4];
        const float* a = asrc + f4 * 4;
        const float* b = adst + f4 * 4;
        s += v.x * a[0] + v.y * a[1] + v.z * a[2] + v.w * a[3];
        t += v.x * b[0] + v.y * b[1] + v.z * b[2] + v.w * b[3];
    }
    as_out[n] = s; ad_out[n] = t;
}

// ================= layer-1 edge kernels (H=4, vectorized per edge) =================

// pass A: e = leaky_relu(a_src[src] + a_dst[dst]); atomic segment max
__global__ void edge_logits4(const int* __restrict__ src, const int* __restrict__ dst,
                             const float* __restrict__ as, const float* __restrict__ ad,
                             float* __restrict__ ebuf, unsigned* __restrict__ mx) {
    unsigned e = blockIdx.x * blockDim.x + threadIdx.x;
    if (e >= NE) return;
    int s = src[e], d = dst[e];
    float4 a = *(const float4*)(as + (size_t)s * 4);
    float4 b = *(const float4*)(ad + (size_t)d * 4);
    float4 v;
    v.x = lrelu(a.x + b.x); v.y = lrelu(a.y + b.y);
    v.z = lrelu(a.z + b.z); v.w = lrelu(a.w + b.w);
    *(float4*)(ebuf + (size_t)e * 4) = v;
    unsigned* m = mx + (size_t)d * 4;
    atomicMax(m + 0, fenc(v.x)); atomicMax(m + 1, fenc(v.y));
    atomicMax(m + 2, fenc(v.z)); atomicMax(m + 3, fenc(v.w));
}

// pass B: ex = exp(e - max[dst]); denom += ex; ebuf := ex
__global__ void edge_denom4(const int* __restrict__ dst, float* __restrict__ ebuf,
                            const unsigned* __restrict__ mx, float* __restrict__ dn) {
    unsigned e = blockIdx.x * blockDim.x + threadIdx.x;
    if (e >= NE) return;
    int d = dst[e];
    float4 v = *(const float4*)(ebuf + (size_t)e * 4);
    uint4  k = *(const uint4*)(mx + (size_t)d * 4);
    float4 ex;
    ex.x = __expf(v.x - fdec(k.x)); ex.y = __expf(v.y - fdec(k.y));
    ex.z = __expf(v.z - fdec(k.z)); ex.w = __expf(v.w - fdec(k.w));
    *(float4*)(ebuf + (size_t)e * 4) = ex;
    float* dp = dn + (size_t)d * 4;
    atomicAdd(dp + 0, ex.x); atomicAdd(dp + 1, ex.y);
    atomicAdd(dp + 2, ex.z); atomicAdd(dp + 3, ex.w);
}

// pass B2: ebuf := ex / (denom + eps) * edge_weight   (final attention coefficient)
__global__ void edge_coef4(const int* __restrict__ dst, const float* __restrict__ ew,
                           float* __restrict__ ebuf, const float* __restrict__ dn) {
    unsigned e = blockIdx.x * blockDim.x + threadIdx.x;
    if (e >= NE) return;
    int d = dst[e];
    float  w  = ew[e];
    float4 ex = *(const float4*)(ebuf + (size_t)e * 4);
    float4 dd = *(const float4*)(dn + (size_t)d * 4);
    float4 c;
    c.x = ex.x / (dd.x + 1e-16f) * w; c.y = ex.y / (dd.y + 1e-16f) * w;
    c.z = ex.z / (dd.z + 1e-16f) * w; c.w = ex.w / (dd.w + 1e-16f) * w;
    *(float4*)(ebuf + (size_t)e * 4) = c;
}

// pass C: out[dst] += coef * h[src]; one thread per (edge, head, f-quad)
__global__ void edge_scatter4(const int* __restrict__ src, const int* __restrict__ dst,
                              const float* __restrict__ coef, const float* __restrict__ h,
                              float* __restrict__ out) {
    unsigned gid = blockIdx.x * blockDim.x + threadIdx.x;
    if (gid >= (unsigned)NE * 32) return;            // 4 heads * 8 quads
    unsigned e  = gid >> 5;
    unsigned r  = gid & 31;
    unsigned hd = r >> 3;
    unsigned q  = r & 7;
    int s = src[e], d = dst[e];
    float c = coef[(size_t)e * 4 + hd];
    unsigned off = hd * HID + q * 4;
    float4 v = *(const float4*)(h + (size_t)s * KDIM + off);
    float* op = out + (size_t)d * KDIM + off;
    atomicAdd(op + 0, c * v.x); atomicAdd(op + 1, c * v.y);
    atomicAdd(op + 2, c * v.z); atomicAdd(op + 3, c * v.w);
}

// ================= layer-2 edge kernels (H=1) =================

__global__ void edge_logits1(const int* __restrict__ src, const int* __restrict__ dst,
                             const float* __restrict__ as, const float* __restrict__ ad,
                             float* __restrict__ ebuf, unsigned* __restrict__ mx) {
    unsigned e = blockIdx.x * blockDim.x + threadIdx.x;
    if (e >= NE) return;
    int s = src[e], d = dst[e];
    float v = lrelu(as[s] + ad[d]);
    ebuf[e] = v;
    atomicMax(mx + d, fenc(v));
}

__global__ void edge_denom1(const int* __restrict__ dst, float* __restrict__ ebuf,
                            const unsigned* __restrict__ mx, float* __restrict__ dn) {
    unsigned e = blockIdx.x * blockDim.x + threadIdx.x;
    if (e >= NE) return;
    int d = dst[e];
    float ex = __expf(ebuf[e] - fdec(mx[d]));
    ebuf[e] = ex;
    atomicAdd(dn + d, ex);
}

__global__ void edge_coef1(const int* __restrict__ dst, const float* __restrict__ ew,
                           float* __restrict__ ebuf, const float* __restrict__ dn) {
    unsigned e = blockIdx.x * blockDim.x + threadIdx.x;
    if (e >= NE) return;
    int d = dst[e];
    ebuf[e] = ebuf[e] / (dn[d] + 1e-16f) * ew[e];
}

// one thread per (edge, f-quad); NCLS = 40 = 10 quads
__global__ void edge_scatter2(const int* __restrict__ src, const int* __restrict__ dst,
                              const float* __restrict__ coef, const float* __restrict__ h,
                              float* __restrict__ out) {
    unsigned gid = blockIdx.x * blockDim.x + threadIdx.x;
    if (gid >= (unsigned)NE * 10) return;
    unsigned e = gid / 10;
    unsigned q = gid % 10;
    int s = src[e], d = dst[e];
    float c = coef[e];
    float4 v = *(const float4*)(h + (size_t)s * NCLS_PAD + q * 4);
    float* op = out + (size_t)d * NCLS + q * 4;
    atomicAdd(op + 0, c * v.x); atomicAdd(op + 1, c * v.y);
    atomicAdd(op + 2, c * v.z); atomicAdd(op + 3, c * v.w);
}

static inline unsigned gridFor(size_t n, int b) { return (unsigned)((n + b - 1) / b); }

extern "C" void kernel_launch(void* const* d_in, const int* in_sizes, int n_in,
                              void* d_out, int out_size, void* d_ws, size_t ws_size,
                              hipStream_t stream) {
    const float* x    = (const float*)d_in[0];
    const int*   ei   = (const int*)d_in[1];
    const float* ew   = (const float*)d_in[2];
    const float* W1   = (const float*)d_in[3];
    const float* aS1  = (const float*)d_in[4];
    const float* aD1  = (const float*)d_in[5];
    const float* b1   = (const float*)d_in[6];
    const float* W2   = (const float*)d_in[7];
    const float* aS2  = (const float*)d_in[8];
    const float* aD2  = (const float*)d_in[9];
    const float* b2   = (const float*)d_in[10];
    float* out        = (float*)d_out;

    const int* src = ei;
    const int* dst = ei + NE;

    // ---- workspace carve (regions reused between layers) ----
    char* w = (char*)d_ws;
    size_t o = 0;
    _Float16* xf16 = (_Float16*)(w + o); o += (size_t)NN * KDIM * 2;        // 25.6 MB (reused as x2 f16)
    _Float16* w1f  = (_Float16*)(w + o); o += 8 * 4 * 32 * 16 * 2;          // 32 KB
    _Float16* w2f  = (_Float16*)(w + o); o += 3 * 4 * 32 * 16 * 2;          // 12 KB
    float*    h1   = (float*)(w + o);    o += (size_t)NN * KDIM * 4;        // 51.2 MB (reused as h2[NN x 48])
    float*    as1  = (float*)(w + o);    o += (size_t)NN * HEADS * 4;       // reused as as2
    float*    ad1  = (float*)(w + o);    o += (size_t)NN * HEADS * 4;       // reused as ad2
    float*    e1   = (float*)(w + o);    o += (size_t)NE * HEADS * 4;       // 25.6 MB (reused as e2)
    unsigned* mx1  = (unsigned*)(w + o); o += (size_t)NN * HEADS * 4;       // reused as mx2
    float*    dn1  = (float*)(w + o);    o += (size_t)NN * HEADS * 4;       // reused as dn2
    float*    out1 = (float*)(w + o);    o += (size_t)NN * KDIM * 4;        // 51.2 MB
    (void)ws_size;

    float*    h2  = h1;            // layer-2 aliases
    float*    as2 = as1; float* ad2 = ad1;
    float*    e2  = e1;  unsigned* mx2 = mx1; float* dn2 = dn1;

    const int B = 256;

    // ---- init accumulators (scratch is poisoned; re-init every call) ----
    zero_u32<<<gridFor((size_t)NN * HEADS, B), B, 0, stream>>>(mx1, NN * HEADS);
    zero_u32<<<gridFor((size_t)NN * HEADS, B), B, 0, stream>>>((unsigned*)dn1, NN * HEADS);
    zero_u32<<<gridFor((size_t)NN * KDIM, B), B, 0, stream>>>((unsigned*)out1, NN * KDIM);

    // ---- layer 1: GEMM (WMMA) ----
    cvt_f32_to_f16<<<gridFor((size_t)NN * KDIM, B), B, 0, stream>>>(x, xf16, NN * KDIM);
    pack_w<<<gridFor(8 * 4 * 32 * 16, B), B, 0, stream>>>(W1, w1f, 128, 8);
    pack_w<<<gridFor(3 * 4 * 32 * 16, B), B, 0, stream>>>(W2, w2f, NCLS, 3);
    {   // 6250 m-tiles x 8 n-tiles, 8 waves/block
        int waves = 6250 * 8;
        gemm_wmma<<<gridFor(waves, 8), 256, 0, stream>>>(xf16, w1f, h1, 6250, 8, KDIM);
    }

    // ---- layer 1: attention ----
    alpha1_kernel<<<gridFor((size_t)NN * HEADS, B), B, 0, stream>>>(h1, aS1, aD1, as1, ad1);
    edge_logits4<<<gridFor(NE, B), B, 0, stream>>>(src, dst, as1, ad1, e1, mx1);
    edge_denom4<<<gridFor(NE, B), B, 0, stream>>>(dst, e1, mx1, dn1);
    edge_coef4<<<gridFor(NE, B), B, 0, stream>>>(dst, ew, e1, dn1);
    edge_scatter4<<<gridFor((size_t)NE * 32, B), B, 0, stream>>>(src, dst, e1, h1, out1);

    // ---- layer 2 input: relu(out1 + b1) -> f16 ----
    relu_bias_cvt<<<gridFor((size_t)NN * KDIM, B), B, 0, stream>>>(out1, b1, xf16, NN * KDIM);

    // ---- layer 2: GEMM (WMMA), N padded 40 -> 48 ----
    {
        int waves = 6250 * 3;
        gemm_wmma<<<gridFor(waves, 8), 256, 0, stream>>>(xf16, w2f, h2, 6250, 3, NCLS_PAD);
    }
    alpha2_kernel<<<gridFor(NN, B), B, 0, stream>>>(h2, aS2, aD2, as2, ad2);

    // re-init softmax state + seed output with bias
    zero_u32<<<gridFor(NN, B), B, 0, stream>>>(mx2, NN);
    zero_u32<<<gridFor(NN, B), B, 0, stream>>>((unsigned*)dn2, NN);
    init_out_bias<<<gridFor((size_t)NN * NCLS, B), B, 0, stream>>>(out, b2, NN * NCLS);

    edge_logits1<<<gridFor(NE, B), B, 0, stream>>>(src, dst, as2, ad2, e2, mx2);
    edge_denom1<<<gridFor(NE, B), B, 0, stream>>>(dst, e2, mx2, dn2);
    edge_coef1<<<gridFor(NE, B), B, 0, stream>>>(dst, ew, e2, dn2);
    edge_scatter2<<<gridFor((size_t)NE * 10, B), B, 0, stream>>>(src, dst, e2, h2, out);
}
